// MultiHeadAttention_27273042330176
// MI455X (gfx1250) — matrix operand order repaired and checked by reading, not compiled, not run on hardware
//
#include <hip/hip_runtime.h>
#include <hip/hip_bf16.h>

// Problem constants (B, T, C) = (2, 256, 384), heads = C heads of dim 1.
#define BB 2
#define TT 256
#define CC 384
#define BTC (BB * TT * CC)   // 196608
#define CxC (CC * CC)        // 147456

typedef __attribute__((ext_vector_type(8)))  _Float16 v8h;
typedef __attribute__((ext_vector_type(16))) _Float16 v16h;
typedef __attribute__((ext_vector_type(8)))  float    v8f;

// ---------------------------------------------------------------------------
// f32 -> f16 conversion
// ---------------------------------------------------------------------------
__global__ __launch_bounds__(256) void cvt_f32_f16(const float* __restrict__ src,
                                                   _Float16* __restrict__ dst, int n) {
    int i = blockIdx.x * 256 + threadIdx.x;
    if (i < n) dst[i] = (_Float16)src[i];
}

// ---------------------------------------------------------------------------
// A-fragment loader for V_WMMA_F32_16X16X32_F16, per ISA layout (7.12.2):
//   lane L (lmod=L%16, lhi=L/16) holds row M=lmod;
//   lhi=0: e0-7 -> K=k0+0..7,  e8-15 -> K=k0+16..23
//   lhi=1: e0-7 -> K=k0+8..15, e8-15 -> K=k0+24..31
// ---------------------------------------------------------------------------
__device__ __forceinline__ v16h load_afrag(const _Float16* __restrict__ row,
                                           int k0, int lhi) {
    v8h alo = *(const v8h*)(row + k0 + lhi * 8);        // 16B aligned
    v8h ahi = *(const v8h*)(row + k0 + 16 + lhi * 8);   // 16B aligned
    return __builtin_shufflevector(alo, ahi,
               0, 1, 2, 3, 4, 5, 6, 7, 8, 9, 10, 11, 12, 13, 14, 15);
}

// ---------------------------------------------------------------------------
// Y[m, n] = sum_k X[m, k] * W[n, k]  (+ bias[n])     i.e.  Y = X @ W^T
// One wave computes a 32x32 output macro-tile: 2 A-frags x 2 B-frags ->
// 4 independent WMMA accumulator chains (good XDL ILP, 2x fragment reuse).
// K = CC = 384 -> 12 k-steps, 48 v_wmma per wave.
// blockIdx.x -> n-macrotile, blockIdx.y -> m-macrotile, blockIdx.z -> batch.
// ---------------------------------------------------------------------------
__global__ __launch_bounds__(32) void gemm_xwt_wmma(const _Float16* __restrict__ Xb,
                                                    const _Float16* __restrict__ Wb,
                                                    float* __restrict__ Yb,
                                                    const float* __restrict__ bias,
                                                    long sX, long sW, long sY) {
    const int z = blockIdx.z;
    const _Float16* X = Xb + (long)z * sX;
    const _Float16* W = Wb + (long)z * sW;
    float*          Y = Yb + (long)z * sY;

    const int n0   = blockIdx.x * 32;
    const int m0   = blockIdx.y * 32;
    const int lane = threadIdx.x;    // 0..31
    const int lmod = lane & 15;      // M (for A), N (for B/C/D)
    const int lhi  = lane >> 4;      // half-wave selector

    const _Float16* xrow0 = X + (long)(m0 + lmod) * CC;       // A rows m0..m0+15
    const _Float16* xrow1 = xrow0 + 16L * CC;                 // A rows m0+16..m0+31
    // B fragment: lane holds column n = row of W; same K order as A fragment
    const _Float16* wrow0 = W + (long)(n0 + lmod) * CC;       // B cols n0..n0+15
    const _Float16* wrow1 = wrow0 + 16L * CC;                 // B cols n0+16..n0+31

    v8f acc00 = {}, acc01 = {}, acc10 = {}, acc11 = {};
#pragma unroll
    for (int k0 = 0; k0 < CC; k0 += 32) {
        v16h a0 = load_afrag(xrow0, k0, lhi);
        v16h a1 = load_afrag(xrow1, k0, lhi);
        v16h b0 = load_afrag(wrow0, k0, lhi);
        v16h b1 = load_afrag(wrow1, k0, lhi);
        acc00 = __builtin_amdgcn_wmma_f32_16x16x32_f16(false, a0, false, b0,
                                                       (short)0, acc00, false, false);
        acc01 = __builtin_amdgcn_wmma_f32_16x16x32_f16(false, a0, false, b1,
                                                       (short)0, acc01, false, false);
        acc10 = __builtin_amdgcn_wmma_f32_16x16x32_f16(false, a1, false, b0,
                                                       (short)0, acc10, false, false);
        acc11 = __builtin_amdgcn_wmma_f32_16x16x32_f16(false, a1, false, b1,
                                                       (short)0, acc11, false, false);
    }

    // C/D layout: VGPR r -> M = tileM + r + 8*lhi, N = tileN + lmod
    const float bv0 = bias ? bias[n0 + lmod]      : 0.0f;
    const float bv1 = bias ? bias[n0 + 16 + lmod] : 0.0f;
#pragma unroll
    for (int r = 0; r < 8; ++r) {
        const int mA = m0 + r + lhi * 8;
        const int mB = mA + 16;
        const int nA = n0 + lmod;
        const int nB = nA + 16;
        Y[(long)mA * CC + nA] = acc00[r] + bv0;
        Y[(long)mA * CC + nB] = acc01[r] + bv1;
        Y[(long)mB * CC + nA] = acc10[r] + bv0;
        Y[(long)mB * CC + nB] = acc11[r] + bv1;
    }
}

// ---------------------------------------------------------------------------
// Per-head (head_dim == 1) causal softmax attention.
// Block = one (h, b); thread i handles query row i.  Two-pass softmax over
// LDS-staged k/v: pass 1 = 4-way independent max reduction, pass 2 = one
// v_exp_f32 per j feeding 4 independent (l, acc) FMA partial sums.
// ---------------------------------------------------------------------------
__global__ __launch_bounds__(256) void attn_head1(const float* __restrict__ q,
                                                  const float* __restrict__ k,
                                                  const float* __restrict__ v,
                                                  _Float16* __restrict__ att) {
    __shared__ float ks[TT];
    __shared__ float vs[TT];

    const int h = blockIdx.x;        // 0..383
    const int b = blockIdx.y;        // 0..1
    const int i = threadIdx.x;       // 0..255

    const long base = (long)b * TT * CC + h;
    ks[i] = k[base + (long)i * CC];
    vs[i] = v[base + (long)i * CC];
    __syncthreads();

    const float scale = 0.05103103630798288f;   // 384^-0.5
    const float qi = q[base + (long)i * CC] * scale;

    const float ninf = -__builtin_inff();

    // Pass 1: m = max_{j<=i} qi*ks[j]  (4 independent chains)
    float m0 = ninf, m1 = ninf, m2 = ninf, m3 = ninf;
    int j = 0;
    for (; j + 3 <= i; j += 4) {
        m0 = fmaxf(m0, qi * ks[j + 0]);
        m1 = fmaxf(m1, qi * ks[j + 1]);
        m2 = fmaxf(m2, qi * ks[j + 2]);
        m3 = fmaxf(m3, qi * ks[j + 3]);
    }
    for (; j <= i; ++j) m0 = fmaxf(m0, qi * ks[j]);
    const float m = fmaxf(fmaxf(m0, m1), fmaxf(m2, m3));

    // Pass 2: l = sum exp(x-m), acc = sum exp(x-m)*v  (4 independent chains)
    float l0 = 0.f, l1 = 0.f, l2 = 0.f, l3 = 0.f;
    float a0 = 0.f, a1 = 0.f, a2 = 0.f, a3 = 0.f;
    j = 0;
    for (; j + 3 <= i; j += 4) {
        const float p0 = __expf(qi * ks[j + 0] - m);
        const float p1 = __expf(qi * ks[j + 1] - m);
        const float p2 = __expf(qi * ks[j + 2] - m);
        const float p3 = __expf(qi * ks[j + 3] - m);
        l0 += p0; a0 = fmaf(p0, vs[j + 0], a0);
        l1 += p1; a1 = fmaf(p1, vs[j + 1], a1);
        l2 += p2; a2 = fmaf(p2, vs[j + 2], a2);
        l3 += p3; a3 = fmaf(p3, vs[j + 3], a3);
    }
    for (; j <= i; ++j) {
        const float p = __expf(qi * ks[j] - m);
        l0 += p; a0 = fmaf(p, vs[j], a0);
    }
    const float l = (l0 + l1) + (l2 + l3);
    const float a = (a0 + a1) + (a2 + a3);
    att[base + (long)i * CC] = (_Float16)(a / l);
}

// ---------------------------------------------------------------------------
// Launch: convert -> fused qkv GEMM (WMMA) -> per-head attention -> proj GEMM
// ---------------------------------------------------------------------------
extern "C" void kernel_launch(void* const* d_in, const int* in_sizes, int n_in,
                              void* d_out, int out_size, void* d_ws, size_t ws_size,
                              hipStream_t stream) {
    const float* q_in = (const float*)d_in[0];
    const float* k_in = (const float*)d_in[1];
    const float* v_in = (const float*)d_in[2];
    // d_in[3] = causal mask (implemented analytically)
    const float* Wq = (const float*)d_in[4];
    const float* Wk = (const float*)d_in[5];
    const float* Wv = (const float*)d_in[6];
    const float* Wp = (const float*)d_in[7];
    const float* bp = (const float*)d_in[8];
    float* out = (float*)d_out;

    // Workspace layout (all sub-buffers stay >=256B aligned):
    //   xh   : 3*BTC f16   (q_in,k_in,v_in in f16)
    //   wh   : 3*CxC f16   (Wq,Wk,Wv in f16)
    //   wph  : CxC  f16    (Wp in f16)
    //   qkv  : 3*BTC f32   (projected q,k,v)
    //   atth : BTC  f16    (attention output)
    _Float16* xh   = (_Float16*)d_ws;
    _Float16* wh   = xh + 3 * (long)BTC;
    _Float16* wph  = wh + 3 * (long)CxC;
    float*    qkv  = (float*)(wph + (long)CxC);
    _Float16* atth = (_Float16*)(qkv + 3 * (long)BTC);

    // 1) f32 -> f16 conversions
    {
        const int nb_x = (BTC + 255) / 256;
        const int nb_w = (CxC + 255) / 256;
        cvt_f32_f16<<<nb_x, 256, 0, stream>>>(q_in, xh + 0 * (long)BTC, BTC);
        cvt_f32_f16<<<nb_x, 256, 0, stream>>>(k_in, xh + 1 * (long)BTC, BTC);
        cvt_f32_f16<<<nb_x, 256, 0, stream>>>(v_in, xh + 2 * (long)BTC, BTC);
        cvt_f32_f16<<<nb_w, 256, 0, stream>>>(Wq, wh + 0 * (long)CxC, CxC);
        cvt_f32_f16<<<nb_w, 256, 0, stream>>>(Wk, wh + 1 * (long)CxC, CxC);
        cvt_f32_f16<<<nb_w, 256, 0, stream>>>(Wv, wh + 2 * (long)CxC, CxC);
        cvt_f32_f16<<<nb_w, 256, 0, stream>>>(Wp, wph, CxC);
    }

    // 2) q/k/v projections: one fused launch, z selects {x, W, y} triple
    {
        dim3 grid(CC / 32, (BB * TT) / 32, 3);   // 12 x 16 x 3
        gemm_xwt_wmma<<<grid, 32, 0, stream>>>(xh, wh, qkv, nullptr,
                                               (long)BTC, (long)CxC, (long)BTC);
    }

    // 3) per-head causal softmax attention (head_dim = 1)
    {
        dim3 grid(CC, BB);                       // 384 x 2
        attn_head1<<<grid, TT, 0, stream>>>(qkv + 0 * (long)BTC,
                                            qkv + 1 * (long)BTC,
                                            qkv + 2 * (long)BTC, atth);
    }

    // 4) output projection + bias -> d_out (f32)
    {
        dim3 grid(CC / 32, (BB * TT) / 32, 1);
        gemm_xwt_wmma<<<grid, 32, 0, stream>>>(atth, wph, out, bp, 0, 0, 0);
    }
}
